// DeeperGCN_21500606284198
// MI455X (gfx1250) — matrix operand order repaired and checked by reading, not compiled, not run on hardware
//
#include <hip/hip_runtime.h>
#include <hip/hip_bf16.h>

// ---------------- problem constants ----------------
#define N_NODES 50000
#define N_EDGES 400000
#define IN_C    32
#define M_C     64
#define N_LAYER 3

typedef __attribute__((ext_vector_type(16))) __bf16 v16bf;
typedef __attribute__((ext_vector_type(8)))  __bf16 v8bf;
typedef __attribute__((ext_vector_type(8)))  float  v8f;

static __device__ __forceinline__ unsigned short f2bf_raw(float f) {
  unsigned u = __float_as_uint(f);
  u += 0x7FFFu + ((u >> 16) & 1u);          // round-to-nearest-even
  return (unsigned short)(u >> 16);
}

static __device__ __forceinline__ float wave_sum(float v) {
#pragma unroll
  for (int o = 16; o > 0; o >>= 1) v += __shfl_xor(v, o, 32);
  return v;
}

static __device__ __forceinline__ v8f splat8(float x) {
  v8f v = {x, x, x, x, x, x, x, x};
  return v;
}

// A fragment (16x32 bf16) from a row-major [16][128] bf16 tile.
// ISA 7.12.2: lane 0-15 row M=lane, halves 0-7 -> K = k0+hi*8+0..7,
// halves 8-15 -> K = k0+hi*8+16..23 (hi = lane>=16).
static __device__ __forceinline__ v16bf load_a_frag(const unsigned short* tile,
                                                    int lane, int k0) {
  const int row = lane & 15;
  const int hi  = (lane >> 4) & 1;
  const unsigned short* rp = tile + row * 128 + k0 + hi * 8;
  v8bf lo = *(const v8bf*)(rp);
  v8bf hh = *(const v8bf*)(rp + 16);
  return __builtin_shufflevector(lo, hh, 0, 1, 2, 3, 4, 5, 6, 7,
                                         8, 9, 10, 11, 12, 13, 14, 15);
}

// B fragment (32x16 bf16) from W^T stored [N][K] row-major bf16.
// Lane: N = lane&15; halves 0-7 -> K = k0 + 8*(lane>=16) + 0..7, halves 8-15 -> +16.
static __device__ __forceinline__ v16bf load_b_frag(const unsigned short* wt, int K,
                                                    int lane, int n0, int k0) {
  const int n    = lane & 15;
  const int koff = k0 + ((lane >> 4) & 1) * 8;
  const unsigned short* bp = wt + (long long)(n0 + n) * K + koff;
  v8bf lo = *(const v8bf*)(bp);
  v8bf hh = *(const v8bf*)(bp + 16);
  return __builtin_shufflevector(lo, hh, 0, 1, 2, 3, 4, 5, 6, 7,
                                         8, 9, 10, 11, 12, 13, 14, 15);
}

static __device__ __forceinline__ v8f wmma_bf16(v16bf a, v16bf b, v8f c) {
  return __builtin_amdgcn_wmma_f32_16x16x32_bf16(false, a, false, b,
                                                 (short)0, c, false, false);
}

// ---------------- small helper kernels ----------------
__global__ void dg_zero_f32(float* __restrict__ p, long long n) {
  long long i = (long long)blockIdx.x * blockDim.x + threadIdx.x;
  if (i < n) p[i] = 0.0f;
}

// transpose+convert: in [L][K][Nc] f32 -> out [L][Nc][K] bf16
__global__ void dg_tconv(const float* __restrict__ in, unsigned short* __restrict__ out,
                         int L, int K, int Nc) {
  long long idx = (long long)blockIdx.x * blockDim.x + threadIdx.x;
  long long tot = (long long)L * K * Nc;
  if (idx >= tot) return;
  int per = K * Nc;
  int l = (int)(idx / per);
  int rem = (int)(idx - (long long)l * per);
  int k = rem / Nc;
  int n = rem - k * Nc;
  out[(long long)l * per + (long long)n * K + k] = f2bf_raw(in[idx]);
}

__global__ void dg_degree(const int* __restrict__ dst, float* __restrict__ deg, int E) {
  int i = blockIdx.x * blockDim.x + threadIdx.x;
  if (i < E) atomicAdd(deg + dst[i], 1.0f);
}

__global__ void dg_invdeg(const float* __restrict__ deg, float* __restrict__ inv, int N) {
  int i = blockIdx.x * blockDim.x + threadIdx.x;
  if (i < N) inv[i] = 1.0f / fmaxf(deg[i], 1.0f);
}

// ---------------- encoder: x@W + b -> LayerNorm -> x_cur (f32) ----------------
__global__ __launch_bounds__(256) void dg_encoder(
    const float* __restrict__ x, const float* __restrict__ W,
    const float* __restrict__ b, const float* __restrict__ g,
    const float* __restrict__ beta, float* __restrict__ xout, int N) {
  const int wid = threadIdx.x >> 5, lane = threadIdx.x & 31;
  const int node = blockIdx.x * 8 + wid;
  if (node >= N) return;
  const float* xr = x + (long long)node * IN_C;
  float a0 = b[lane], a1 = b[lane + 32];
#pragma unroll
  for (int k = 0; k < IN_C; ++k) {
    float xv = xr[k];
    a0 += xv * W[k * M_C + lane];
    a1 += xv * W[k * M_C + lane + 32];
  }
  float mean = wave_sum(a0 + a1) * (1.0f / 64.0f);
  float d0 = a0 - mean, d1 = a1 - mean;
  float var = wave_sum(d0 * d0 + d1 * d1) * (1.0f / 64.0f);
  float rstd = rsqrtf(var + 1e-5f);
  long long o = (long long)node * M_C;
  xout[o + lane]      = d0 * rstd * g[lane] + beta[lane];
  xout[o + lane + 32] = d1 * rstd * g[lane + 32] + beta[lane + 32];
}

// ---------------- pre-norm: h = relu(LN(x)) -> bf16 ----------------
__global__ __launch_bounds__(256) void dg_prenorm(
    const float* __restrict__ xin, const float* __restrict__ g,
    const float* __restrict__ b, unsigned short* __restrict__ hout, int N) {
  const int wid = threadIdx.x >> 5, lane = threadIdx.x & 31;
  const int node = blockIdx.x * 8 + wid;
  if (node >= N) return;
  long long o = (long long)node * M_C;
  float a0 = xin[o + lane], a1 = xin[o + lane + 32];
  float mean = wave_sum(a0 + a1) * (1.0f / 64.0f);
  float d0 = a0 - mean, d1 = a1 - mean;
  float var = wave_sum(d0 * d0 + d1 * d1) * (1.0f / 64.0f);
  float rstd = rsqrtf(var + 1e-5f);
  float y0 = fmaxf(d0 * rstd * g[lane] + b[lane], 0.0f);
  float y1 = fmaxf(d1 * rstd * g[lane + 32] + b[lane + 32], 0.0f);
  hout[o + lane]      = f2bf_raw(y0);
  hout[o + lane + 32] = f2bf_raw(y1);
}

// ---------------- edge MLP (the big WMMA kernel) ----------------
// msg = relu(concat(h[dst],h[src]) @ W1 + b1) @ W2 + b2 ; atomicAdd into agg[dst]
// LDS: W1T bf16 128x128 | W2T bf16 64x128 | b1 f32 128 | b2 f32 64 |
//      per-wave: aT bf16 16x128 + hT bf16 16x128
#define EDGE_SMEM_BYTES ((128*128 + 64*128) * 2 + (128 + 64) * 4 + 8 * 2 * 16 * 128 * 2)
#define EDGE_BLOCKS 784

__global__ __launch_bounds__(256) void dg_edge_mlp(
    const int* __restrict__ srcIdx, const int* __restrict__ dstIdx,
    const unsigned short* __restrict__ hbf,
    const unsigned short* __restrict__ W1T, const unsigned short* __restrict__ W2T,
    const float* __restrict__ b1, const float* __restrict__ b2,
    float* __restrict__ agg, int E) {
  extern __shared__ char smem[];
  unsigned short* sW1 = (unsigned short*)smem;          // 128*128
  unsigned short* sW2 = sW1 + 128 * 128;                // 64*128
  float* sb1 = (float*)(sW2 + 64 * 128);
  float* sb2 = sb1 + 128;
  unsigned short* warp_base = (unsigned short*)(sb2 + 64);

  const int tid = threadIdx.x;
  const int wid = tid >> 5;
  const int lane = tid & 31;

  { // cooperative weight staging (once per block, amortized over the tile loop)
    const uint4* g1 = (const uint4*)W1T;
    uint4* s1 = (uint4*)sW1;
    for (int i = tid; i < (128 * 128) / 8; i += 256) s1[i] = g1[i];
    const uint4* g2 = (const uint4*)W2T;
    uint4* s2 = (uint4*)sW2;
    for (int i = tid; i < (64 * 128) / 8; i += 256) s2[i] = g2[i];
    if (tid < 128) sb1[tid] = b1[tid];
    if (tid < 64)  sb2[tid] = b2[tid];
  }
  __syncthreads();

  unsigned short* aT = warp_base + wid * (2 * 16 * 128);
  unsigned short* hT = aT + 16 * 128;

  const int ntiles = (E + 15) >> 4;
  const int hi = lane >> 4;
  const int cl = lane & 15;

  // grid-stride loop over 16-edge tiles (wave-uniform)
  for (int tile = blockIdx.x * 8 + wid; tile < ntiles; tile += gridDim.x * 8) {

    { // gather concat(h[dst], h[src]) into aT (bf16, row per edge)
      const int r = lane & 15;
      const int sel = lane >> 4;        // 0 -> x_i = h[dst], 1 -> x_j = h[src]
      const int e = tile * 16 + r;
      const bool valid = e < E;
      int node = 0;
      if (valid) node = sel ? srcIdx[e] : dstIdx[e];
      const uint4* sp = (const uint4*)(hbf + (long long)node * M_C);
      uint4* dp = (uint4*)(aT + r * 128 + sel * 64);
#pragma unroll
      for (int j = 0; j < 8; ++j) {
        uint4 v = make_uint4(0u, 0u, 0u, 0u);
        if (valid) v = sp[j];
        dp[j] = v;
      }
    }
    // hoist destination indices for the scatter (one load per row)
    int drow[8];
#pragma unroll
    for (int r = 0; r < 8; ++r) {
      int e = tile * 16 + r + hi * 8;
      drow[r] = (e < E) ? dstIdx[e] : -1;
    }
    asm volatile("s_wait_dscnt 0" ::: "memory");

    // GEMM1: [16x128] @ W1 [128x128] + b1, ReLU -> hT (bf16)
    for (int nt = 0; nt < 8; ++nt) {
      v8f acc = splat8(sb1[nt * 16 + cl]);
#pragma unroll
      for (int kt = 0; kt < 4; ++kt) {
        v16bf a = load_a_frag(aT, lane, kt * 32);
        v16bf b = load_b_frag(sW1, 128, lane, nt * 16, kt * 32);
        acc = wmma_bf16(a, b, acc);
      }
#pragma unroll
      for (int r = 0; r < 8; ++r) {
        float v = fmaxf(acc[r], 0.0f);   // single v_max_num_f32
        hT[(r + hi * 8) * 128 + (nt * 16 + cl)] = f2bf_raw(v);
      }
    }
    asm volatile("s_wait_dscnt 0" ::: "memory");

    // GEMM2: [16x128] @ W2 [128x64] + b2, scatter-add into agg[dst]
    for (int nt = 0; nt < 4; ++nt) {
      v8f acc = splat8(sb2[nt * 16 + cl]);
#pragma unroll
      for (int kt = 0; kt < 4; ++kt) {
        v16bf a = load_a_frag(hT, lane, kt * 32);
        v16bf b = load_b_frag(sW2, 128, lane, nt * 16, kt * 32);
        acc = wmma_bf16(a, b, acc);
      }
#pragma unroll
      for (int r = 0; r < 8; ++r) {
        if (drow[r] >= 0) {
          atomicAdd(agg + (long long)drow[r] * M_C + nt * 16 + cl, acc[r]);
        }
      }
    }
  }
}

// ---------------- node update: x_out = x_in + agg*inv_deg + h@Wr (WMMA) -------
__global__ __launch_bounds__(256) void dg_node_update(
    const float* __restrict__ xin, const float* __restrict__ agg,
    const float* __restrict__ inv_deg, const unsigned short* __restrict__ hbf,
    const unsigned short* __restrict__ WrT, float* __restrict__ xout, int N) {
  __shared__ unsigned short sWr[64 * 64];
  const int tid = threadIdx.x, wid = tid >> 5, lane = tid & 31;
  {
    const uint4* g = (const uint4*)WrT;
    uint4* s = (uint4*)sWr;
    for (int i = tid; i < (64 * 64) / 8; i += 256) s[i] = g[i];
  }
  __syncthreads();

  const int ntiles = (N + 15) >> 4;
  const int tile = blockIdx.x * 8 + wid;
  if (tile >= ntiles) return;

  const int hi = lane >> 4, cl = lane & 15;
  int arow = tile * 16 + cl;
  if (arow >= N) arow = N - 1;
  const unsigned short* rp = hbf + (long long)arow * M_C;
  v8bf a0lo = *(const v8bf*)(rp + hi * 8);
  v8bf a0hi = *(const v8bf*)(rp + 16 + hi * 8);
  v16bf a0 = __builtin_shufflevector(a0lo, a0hi, 0, 1, 2, 3, 4, 5, 6, 7,
                                                 8, 9, 10, 11, 12, 13, 14, 15);
  v8bf a1lo = *(const v8bf*)(rp + 32 + hi * 8);
  v8bf a1hi = *(const v8bf*)(rp + 48 + hi * 8);
  v16bf a1 = __builtin_shufflevector(a1lo, a1hi, 0, 1, 2, 3, 4, 5, 6, 7,
                                                 8, 9, 10, 11, 12, 13, 14, 15);
  // hoist per-row validity and 1/deg out of the nt loop
  bool  rok[8];
  float rinv[8];
#pragma unroll
  for (int r = 0; r < 8; ++r) {
    int row = tile * 16 + r + hi * 8;
    rok[r] = row < N;
    rinv[r] = rok[r] ? inv_deg[row] : 0.0f;
  }
  for (int nt = 0; nt < 4; ++nt) {
    v8f acc = splat8(0.0f);
    acc = wmma_bf16(a0, load_b_frag(sWr, 64, lane, nt * 16, 0), acc);
    acc = wmma_bf16(a1, load_b_frag(sWr, 64, lane, nt * 16, 32), acc);
#pragma unroll
    for (int r = 0; r < 8; ++r) {
      if (rok[r]) {
        long long idx = (long long)(tile * 16 + r + hi * 8) * M_C + nt * 16 + cl;
        xout[idx] = xin[idx] + agg[idx] * rinv[r] + acc[r];
      }
    }
  }
}

// ---------------- host launcher ----------------
extern "C" void kernel_launch(void* const* d_in, const int* in_sizes, int n_in,
                              void* d_out, int out_size, void* d_ws, size_t ws_size,
                              hipStream_t stream) {
  (void)n_in; (void)out_size; (void)ws_size;
  const float* x      = (const float*)d_in[0];
  const int*   eidx   = (const int*)d_in[1];   // jax default x64 off -> int32
  const float* enc_W  = (const float*)d_in[2];
  const float* enc_b  = (const float*)d_in[3];
  const float* enc_g  = (const float*)d_in[4];
  const float* enc_be = (const float*)d_in[5];
  const float* ln_g   = (const float*)d_in[6];
  const float* ln_b   = (const float*)d_in[7];
  const float* W1     = (const float*)d_in[8];
  const float* b1     = (const float*)d_in[9];
  const float* W2     = (const float*)d_in[10];
  const float* b2     = (const float*)d_in[11];
  const float* Wr     = (const float*)d_in[12];

  const int N = in_sizes[0] / IN_C;
  const int E = in_sizes[1] / 2;
  const int* srcIdx = eidx;          // edge_index[0]
  const int* dstIdx = eidx + E;      // edge_index[1]

  // workspace layout
  char* ws = (char*)d_ws;
  float* x_cur = (float*)(ws);                               // N*64 f32
  float* agg   = (float*)(ws + (size_t)N * 64 * 4);          // N*64 f32
  unsigned short* hbf = (unsigned short*)(ws + (size_t)N * 64 * 8);      // N*64 bf16
  float* deg   = (float*)(ws + (size_t)N * 64 * 8 + (size_t)N * 64 * 2);
  float* invd  = deg + N;
  unsigned short* W1T = (unsigned short*)(invd + N);                     // 3*128*128
  unsigned short* W2T = W1T + 3 * 128 * 128;                             // 3*64*128
  unsigned short* WrT = W2T + 3 * 64 * 128;                              // 3*64*64

  // --- weight prep (transpose + bf16) ---
  dg_tconv<<<(3 * 128 * 128 + 255) / 256, 256, 0, stream>>>(W1, W1T, 3, 128, 128);
  dg_tconv<<<(3 * 128 * 64 + 255) / 256, 256, 0, stream>>>(W2, W2T, 3, 128, 64);
  dg_tconv<<<(3 * 64 * 64 + 255) / 256, 256, 0, stream>>>(Wr, WrT, 3, 64, 64);

  // --- encoder ---
  dg_encoder<<<(N + 7) / 8, 256, 0, stream>>>(x, enc_W, enc_b, enc_g, enc_be, x_cur, N);

  // --- degree / inverse degree ---
  dg_zero_f32<<<(N + 255) / 256, 256, 0, stream>>>(deg, N);
  dg_degree<<<(E + 255) / 256, 256, 0, stream>>>(dstIdx, deg, E);
  dg_invdeg<<<(N + 255) / 256, 256, 0, stream>>>(deg, invd, N);

  const int nblocks = (((N + 15) / 16) + 7) / 8;

  for (int l = 0; l < N_LAYER; ++l) {
    dg_prenorm<<<(N + 7) / 8, 256, 0, stream>>>(x_cur, ln_g + l * 64, ln_b + l * 64, hbf, N);
    dg_zero_f32<<<((long long)N * 64 + 255) / 256, 256, 0, stream>>>(agg, (long long)N * 64);
    dg_edge_mlp<<<EDGE_BLOCKS, 256, EDGE_SMEM_BYTES, stream>>>(
        srcIdx, dstIdx, hbf,
        W1T + (size_t)l * 128 * 128, W2T + (size_t)l * 64 * 128,
        b1 + l * 128, b2 + l * 64, agg, E);
    float* xout = (l == N_LAYER - 1) ? (float*)d_out : x_cur;
    dg_node_update<<<nblocks, 256, 0, stream>>>(
        x_cur, agg, invd, hbf, WrT + (size_t)l * 64 * 64, xout, N);
  }
}
